// MinimalLightningAttention2_80058190398041
// MI455X (gfx1250) — matrix operand order repaired and checked by reading, not compiled, not run on hardware
//
#include <hip/hip_runtime.h>

// ---------------------------------------------------------------------------
// Types for CDNA5 WMMA (wave32): v_wmma_f32_16x16x32_bf16
// ---------------------------------------------------------------------------
typedef __bf16 bf16_t;
typedef __attribute__((ext_vector_type(16))) __bf16 v16bf;
typedef __attribute__((ext_vector_type(8)))  __bf16 v8bf;
typedef __attribute__((ext_vector_type(8)))  float  v8f;

// Problem constants (from reference setup_inputs)
#define BATCH   2
#define SEQ     4096
#define CDIM    2048
#define NHEAD   16
#define HDIM    128
#define CHUNK   64
#define NCHUNK  (SEQ / CHUNK)
#define QKVN    (3 * CDIM)          // 6144
#define MROWS   (BATCH * SEQ)       // 8192

// ---------------------------------------------------------------------------
// CDNA5 async global->LDS copy (tracked by ASYNCcnt) + wait, via inline asm
// (CDNA5 ISA GLOBAL_LOAD_ASYNC_TO_LDS_B128, CDNA5 ISA)
// ---------------------------------------------------------------------------
__device__ __forceinline__ void async_ld_lds_b128(void* lds, const void* gptr) {
  unsigned l = (unsigned)(uintptr_t)lds;          // LDS byte address (addr[31:0])
  asm volatile("global_load_async_to_lds_b128 %0, %1, off"
               :: "v"(l), "v"(gptr) : "memory");
}
#define WAIT_ASYNC(n) asm volatile("s_wait_asynccnt %0" :: "n"(n) : "memory")

// ---------------------------------------------------------------------------
// Fragment loaders (wave32 WMMA layouts, CDNA5 ISA §7.12.2)
// A 16-bit 16x32: lane m=l&15; code's own k order (not ISA table), 16 contiguous K elems at (l>=16)*16 (matches B).
// B 16-bit 32x16: lane n=l&15; same k order as A, 16 contiguous K elems at (l>=16)*16 -> store
//   B transposed [n][k] so the load is contiguous.
// C/D f32 16x16: lane n=l&15, row m=(l>=16?8:0)+vgpr.
// ---------------------------------------------------------------------------
__device__ __forceinline__ v16bf load_a_frag(const bf16_t* rowk, int lane) {
  const int hi8 = (lane & 16) ? 16 : 0;
  v8bf lo = *(const v8bf*)(rowk + hi8);
  v8bf hi = *(const v8bf*)(rowk + 8 + hi8);
  v16bf r;
#pragma unroll
  for (int i = 0; i < 8; ++i) { r[i] = lo[i]; r[i + 8] = hi[i]; }
  return r;
}

__device__ __forceinline__ v16bf load_b_frag(const bf16_t* p) {
  v8bf lo = *(const v8bf*)(p);
  v8bf hi = *(const v8bf*)(p + 8);
  v16bf r;
#pragma unroll
  for (int i = 0; i < 8; ++i) { r[i] = lo[i]; r[i + 8] = hi[i]; }
  return r;
}

__device__ __forceinline__ v8f wmma_bf16(v16bf a, v16bf b, v8f c) {
  return __builtin_amdgcn_wmma_f32_16x16x32_bf16(
      false, a, false, b, (short)0, c, false, false);
}

// ---------------------------------------------------------------------------
// Prep kernels: fp32 -> bf16 convert, and convert+transpose for weights.
// ---------------------------------------------------------------------------
__global__ __launch_bounds__(256) void cvt_f32_to_bf16(
    const float* __restrict__ in, bf16_t* __restrict__ out, size_t n) {
  size_t i = ((size_t)blockIdx.x * 256 + threadIdx.x) * 4;
  if (i + 3 < n) {
#pragma unroll
    for (int j = 0; j < 4; ++j) out[i + j] = (bf16_t)in[i + j];
  }
}

// W[K][N] fp32 -> Wt[N][K] bf16 (32x32 tiles via LDS)
__global__ __launch_bounds__(256) void transpose_cvt_bf16(
    const float* __restrict__ W, bf16_t* __restrict__ Wt, int K, int N) {
  __shared__ float tile[32][33];
  const int t = threadIdx.x;
  const int tx = t & 31, ty = t >> 5;           // 32 x 8
  const int k0 = blockIdx.y * 32, n0 = blockIdx.x * 32;
#pragma unroll
  for (int i = 0; i < 32; i += 8)
    tile[ty + i][tx] = W[(size_t)(k0 + ty + i) * N + (n0 + tx)];
  __syncthreads();
#pragma unroll
  for (int i = 0; i < 32; i += 8)
    Wt[(size_t)(n0 + ty + i) * K + (k0 + tx)] = (bf16_t)tile[tx][ty + i];
}

// ---------------------------------------------------------------------------
// Tiled GEMM:  C[M,N] = A[M,K] * Wt[N,K]^T + bias[N]
// A, Wt bf16 row-major (K contiguous): both tiles staged with async b128
// copies into double-buffered LDS; WMMA on the current buffer overlaps the
// async fill of the next.  WG tile 128x256, 8 waves in a 2x4 grid, each wave
// computes 64x64 = 4x4 WMMA tiles (16 f32x8 accumulators -> 4x fragment
// reuse on both operands, 16 back-to-back WMMAs per K-step).
// ---------------------------------------------------------------------------
#define LDS_STRIDE 40   // 32 K elems + 8 pad (80B rows, 16B aligned)

template <int CHUNKS>
__device__ __forceinline__ void stage_tile_async(
    bf16_t* __restrict__ sDst, const bf16_t* __restrict__ gSrc,
    int ld_g, int row0, int k0, int t) {
  // CHUNKS*256 chunks of 16B cover (CHUNKS*64) rows x 32 cols bf16
#pragma unroll
  for (int i = 0; i < CHUNKS; ++i) {
    int ch = i * 256 + t;
    int r = ch >> 2, c8 = (ch & 3) * 8;
    async_ld_lds_b128(sDst + r * LDS_STRIDE + c8,
                      gSrc + (size_t)(row0 + r) * ld_g + k0 + c8);
  }
}

template <bool OUT_BF16>
__global__ __launch_bounds__(256) void gemm_bf16_wmma(
    const bf16_t* __restrict__ A, const bf16_t* __restrict__ Wt,
    const float* __restrict__ bias, void* __restrict__ Cptr,
    int M, int N, int K) {
  __shared__ __align__(16) bf16_t sA[2][128 * LDS_STRIDE];  // 20 KB
  __shared__ __align__(16) bf16_t sB[2][256 * LDS_STRIDE];  // 40 KB

  const int t = threadIdx.x;
  const int lane = t & 31;
  const int w = t >> 5;
  const int waveM = w >> 2;   // 0..1  (64 rows each)
  const int waveN = w & 3;    // 0..3  (64 cols each)
  const int tileM = blockIdx.y * 128;
  const int tileN = blockIdx.x * 256;
  const int n_lo = lane & 15;
  const int kb = (lane & 16) ? 16 : 0;

  v8f acc[4][4] = {};

  const int TK = K >> 5;
  // prologue: fill buffer 0 (A: 2 chunks/thread, B: 4 chunks/thread)
  stage_tile_async<2>(sA[0], A, K, tileM, 0, t);
  stage_tile_async<4>(sB[0], Wt, K, tileN, 0, t);

  for (int i = 0; i < TK; ++i) {
    const int cur = i & 1;
    if (i + 1 < TK) {
      stage_tile_async<2>(sA[cur ^ 1], A, K, tileM, (i + 1) << 5, t);
      stage_tile_async<4>(sB[cur ^ 1], Wt, K, tileN, (i + 1) << 5, t);
      WAIT_ASYNC(6);        // our 6 copies for tile i done (in-order returns)
    } else {
      WAIT_ASYNC(0);
    }
    __syncthreads();        // everyone's tile-i copies visible

    const bf16_t* sAc = sA[cur];
    const bf16_t* sBc = sB[cur];
    v16bf bfrag[4];
#pragma unroll
    for (int nt = 0; nt < 4; ++nt)
      bfrag[nt] =
          load_b_frag(sBc + (waveN * 64 + nt * 16 + n_lo) * LDS_STRIDE + kb);
#pragma unroll
    for (int mt = 0; mt < 4; ++mt) {
      v16bf afrag =
          load_a_frag(sAc + (waveM * 64 + mt * 16 + n_lo) * LDS_STRIDE, lane);
#pragma unroll
      for (int nt = 0; nt < 4; ++nt)
        acc[mt][nt] = wmma_bf16(afrag, bfrag[nt], acc[mt][nt]);
    }
    __syncthreads();        // all reads of buf[cur] done before it refills
  }

  const int m_hi = (lane & 16) ? 8 : 0;
#pragma unroll
  for (int mt = 0; mt < 4; ++mt) {
#pragma unroll
    for (int nt = 0; nt < 4; ++nt) {
      int col = tileN + waveN * 64 + nt * 16 + n_lo;
      float bv = bias[col];
#pragma unroll
      for (int v = 0; v < 8; ++v) {
        int row = tileM + waveM * 64 + mt * 16 + m_hi + v;
        float val = acc[mt][nt][v] + bv;
        if (OUT_BF16)
          ((bf16_t*)Cptr)[(size_t)row * N + col] = (bf16_t)val;
        else
          ((float*)Cptr)[(size_t)row * N + col] = val;
      }
    }
  }
}

// ---------------------------------------------------------------------------
// Lightning attention: one WG (256 thr, 8 waves) per (batch, head).
// Per 64-row chunk: q/k/v tiles stream in via async-to-LDS b128; S->Sbt bf16
// rebuild overlaps the async copies; then three WMMA passes:
//   scores = (q kT) * intra_decay            (64x64)
//   o      = scores v + q_decay * (q S)      (64x128)
//   S      = blk_decay*S + (k*k_decay)T v    (128x128)
// ---------------------------------------------------------------------------
__global__ __launch_bounds__(256) void lightning_attn_wmma(
    const bf16_t* __restrict__ qkv,   // [B][SEQ][6144] bf16
    const float* __restrict__ slopes, // [16]
    bf16_t* __restrict__ attn) {      // [B][SEQ][2048] bf16 (B,N,H*hd)
  extern __shared__ __align__(16) char smem[];
  float*  sS   = (float*) (smem);               // 128x128 f32   (65536 B)
  bf16_t* sSbt = (bf16_t*)(smem + 65536);       // Sbt[e][d]     (32768 B)
  bf16_t* sQ   = (bf16_t*)(smem + 98304);       // q [i][d]      (16384 B)
  bf16_t* sK   = (bf16_t*)(smem + 114688);      // k [j][d]      (16384 B)
  bf16_t* sV   = (bf16_t*)(smem + 131072);      // v [j][e]      (16384 B)
  bf16_t* sKdT = (bf16_t*)(smem + 147456);      // (k*kdec)T[d][j](16384 B)
  bf16_t* sVt  = (bf16_t*)(smem + 163840);      // vT [e][j]     (16384 B)
  bf16_t* sSc  = (bf16_t*)(smem + 180224);      // scores [i][j] ( 8192 B)

  const int t = threadIdx.x;
  const int lane = t & 31;
  const int w = t >> 5;
  const int h = blockIdx.x & (NHEAD - 1);
  const int b = blockIdx.x >> 4;

  const float s = slopes[h];
  const float blk_decay = __expf(-s * (float)CHUNK);

  for (int i = t; i < HDIM * HDIM; i += 256) sS[i] = 0.f;
  __syncthreads();

  const bf16_t* qbase = qkv + (size_t)b * SEQ * QKVN + (size_t)h * HDIM;
  const int n_lo = lane & 15;
  const int m_hi = (lane & 16) ? 8 : 0;
  const int kb = (lane & 16) ? 16 : 0;

  for (int c = 0; c < NCHUNK; ++c) {
    const size_t row0 = (size_t)c * CHUNK;

    // ---- async-stage q, k, v tiles (12 b128 copies per wave) ----
#pragma unroll
    for (int i = 0; i < 4; ++i) {
      int ch = i * 256 + t;             // 0..1023 over 64x(128/8)
      int r = ch >> 4, c8 = (ch & 15) * 8;
      const bf16_t* gp = qbase + (row0 + r) * QKVN + c8;
      async_ld_lds_b128(sQ + r * HDIM + c8, gp);
      async_ld_lds_b128(sK + r * HDIM + c8, gp + CDIM);
      async_ld_lds_b128(sV + r * HDIM + c8, gp + 2 * CDIM);
    }
    // ---- overlap: rebuild Sbt[e][d] = bf16(S[d][e]) while copies fly ----
#pragma unroll 4
    for (int i = 0; i < 64; ++i) {
      int idx = i * 256 + t;            // 0..16383
      int e = idx >> 7, d = idx & 127;
      sSbt[e * HDIM + d] = (bf16_t)sS[d * HDIM + e];
    }
    WAIT_ASYNC(0);
    __syncthreads();

    // ---- LDS->LDS transposes: kdT (with decay) and vT ----
#pragma unroll 4
    for (int i = 0; i < 32; ++i) {
      int idx = i * 256 + t;            // 0..8191
      int r = idx >> 7, d = idx & 127;
      float kv = (float)sK[r * HDIM + d];
      sKdT[d * CHUNK + r] = (bf16_t)(kv * __expf(-s * (float)(CHUNK - r)));
      sVt[d * CHUNK + r] = sV[r * HDIM + d];
    }
    __syncthreads();

    // ---- pass 1: scores = q kT with intra decay; wave = 32x16 region ----
    {
      const int Mb = (w >> 2) * 32;
      const int Nb = (w & 3) * 16;
      v8f accs[2] = {};
      const int j = Nb + n_lo;
#pragma unroll
      for (int kc = 0; kc < 4; ++kc) {  // K = 128
        v16bf bfrag = load_b_frag(sK + j * HDIM + kc * 32 + kb);
#pragma unroll
        for (int mt = 0; mt < 2; ++mt) {
          v16bf afrag =
              load_a_frag(sQ + (Mb + mt * 16 + n_lo) * HDIM + kc * 32, lane);
          accs[mt] = wmma_bf16(afrag, bfrag, accs[mt]);
        }
      }
#pragma unroll
      for (int mt = 0; mt < 2; ++mt)
#pragma unroll
        for (int v = 0; v < 8; ++v) {
          int i_ = Mb + mt * 16 + m_hi + v;
          float dec = (i_ >= j) ? __expf(-s * (float)(i_ - j)) : 0.f;
          sSc[i_ * CHUNK + j] = (bf16_t)(accs[mt][v] * dec);
        }
    }
    __syncthreads();

    // ---- pass 2: o = scores*v + qdec*(q*S); wave = 32x32 region ----
    {
      const int Mb = (w >> 2) * 32;
      const int Nb = (w & 3) * 32;
      v8f accO[2][2] = {};
      v8f accI[2][2] = {};
#pragma unroll
      for (int kc = 0; kc < 2; ++kc) {  // o_intra: K = 64
        v16bf b0 = load_b_frag(sVt + (Nb + n_lo) * CHUNK + kc * 32 + kb);
        v16bf b1 = load_b_frag(sVt + (Nb + 16 + n_lo) * CHUNK + kc * 32 + kb);
#pragma unroll
        for (int mt = 0; mt < 2; ++mt) {
          v16bf a =
              load_a_frag(sSc + (Mb + mt * 16 + n_lo) * CHUNK + kc * 32, lane);
          accO[mt][0] = wmma_bf16(a, b0, accO[mt][0]);
          accO[mt][1] = wmma_bf16(a, b1, accO[mt][1]);
        }
      }
#pragma unroll
      for (int kc = 0; kc < 4; ++kc) {  // o_inter: K = 128
        v16bf b0 = load_b_frag(sSbt + (Nb + n_lo) * HDIM + kc * 32 + kb);
        v16bf b1 = load_b_frag(sSbt + (Nb + 16 + n_lo) * HDIM + kc * 32 + kb);
#pragma unroll
        for (int mt = 0; mt < 2; ++mt) {
          v16bf a =
              load_a_frag(sQ + (Mb + mt * 16 + n_lo) * HDIM + kc * 32, lane);
          accI[mt][0] = wmma_bf16(a, b0, accI[mt][0]);
          accI[mt][1] = wmma_bf16(a, b1, accI[mt][1]);
        }
      }
      bf16_t* obase = attn + ((size_t)b * SEQ + row0) * CDIM + (size_t)h * HDIM;
#pragma unroll
      for (int mt = 0; mt < 2; ++mt)
#pragma unroll
        for (int v = 0; v < 8; ++v) {
          int i_ = Mb + mt * 16 + m_hi + v;
          float qdec = __expf(-s * (float)i_);
#pragma unroll
          for (int nt = 0; nt < 2; ++nt) {
            int e = Nb + nt * 16 + n_lo;
            float val = accO[mt][nt][v] + qdec * accI[mt][nt][v];
            obase[(size_t)i_ * CDIM + e] = (bf16_t)val;
          }
        }
    }

    // ---- pass 3: S = blk_decay*S + kdT * v; wave = 64x32 region ----
    {
      const int Mb = (w >> 2) * 64;
      const int Nb = (w & 3) * 32;
#pragma unroll
      for (int mt = 0; mt < 4; ++mt) {
        v16bf a0 = load_a_frag(sKdT + (Mb + mt * 16 + n_lo) * CHUNK + 0, lane);
        v16bf a1 = load_a_frag(sKdT + (Mb + mt * 16 + n_lo) * CHUNK + 32, lane);
#pragma unroll
        for (int nt = 0; nt < 2; ++nt) {
          int e = Nb + nt * 16 + n_lo;
          v8f accS = {};
#pragma unroll
          for (int v = 0; v < 8; ++v)
            accS[v] = blk_decay * sS[(Mb + mt * 16 + m_hi + v) * HDIM + e];
          v16bf b0 = load_b_frag(sVt + (Nb + nt * 16 + n_lo) * CHUNK + kb);
          v16bf b1 = load_b_frag(sVt + (Nb + nt * 16 + n_lo) * CHUNK + 32 + kb);
          accS = wmma_bf16(a0, b0, accS);
          accS = wmma_bf16(a1, b1, accS);
#pragma unroll
          for (int v = 0; v < 8; ++v)
            sS[(Mb + mt * 16 + m_hi + v) * HDIM + e] = accS[v];
        }
      }
    }
    __syncthreads();  // S final before next chunk reads it
  }
}

// ---------------------------------------------------------------------------
// Host-side launch
// ---------------------------------------------------------------------------
extern "C" void kernel_launch(void* const* d_in, const int* in_sizes, int n_in,
                              void* d_out, int out_size, void* d_ws,
                              size_t ws_size, hipStream_t stream) {
  const float* x      = (const float*)d_in[0];
  const float* Wqkv   = (const float*)d_in[1];
  const float* bqkv   = (const float*)d_in[2];
  const float* Wout   = (const float*)d_in[3];
  const float* bout   = (const float*)d_in[4];
  const float* slopes = (const float*)d_in[5];
  float* out = (float*)d_out;

  // workspace carve (bytes)
  char* ws = (char*)d_ws;
  bf16_t* qkv_bf  = (bf16_t*)(ws);                       // 8192*6144*2 = 100663296
  bf16_t* attn_bf = (bf16_t*)(ws + 100663296ULL);        // 8192*2048*2 =  33554432
  bf16_t* x_bf    = (bf16_t*)(ws + 134217728ULL);        // 8192*2048*2 =  33554432
  bf16_t* Wqkv_t  = (bf16_t*)(ws + 167772160ULL);        // 6144*2048*2 =  25165824
  bf16_t* Wout_t  = (bf16_t*)(ws + 192937984ULL);        // 2048*2048*2 =   8388608

  // 0) prep: bf16 conversions (+ weight transposes to N x K layout)
  cvt_f32_to_bf16<<<(MROWS * CDIM) / (256 * 4), 256, 0, stream>>>(
      x, x_bf, (size_t)MROWS * CDIM);
  transpose_cvt_bf16<<<dim3(QKVN / 32, CDIM / 32), 256, 0, stream>>>(
      Wqkv, Wqkv_t, CDIM, QKVN);
  transpose_cvt_bf16<<<dim3(CDIM / 32, CDIM / 32), 256, 0, stream>>>(
      Wout, Wout_t, CDIM, CDIM);

  // 1) qkv = x @ Wqkv + bqkv -> bf16
  dim3 g1(QKVN / 256, MROWS / 128);
  gemm_bf16_wmma<true><<<g1, 256, 0, stream>>>(
      x_bf, Wqkv_t, bqkv, qkv_bf, MROWS, QKVN, CDIM);

  // 2) chunked lightning attention (one WG per (b,h))
  size_t smem = 188416;
  lightning_attn_wmma<<<dim3(BATCH * NHEAD), 256, smem, stream>>>(
      qkv_bf, slopes, attn_bf);

  // 3) out = attn @ Wout + bout -> fp32
  dim3 g2(CDIM / 256, MROWS / 128);
  gemm_bf16_wmma<false><<<g2, 256, 0, stream>>>(
      attn_bf, Wout_t, bout, out, MROWS, CDIM, CDIM);
}
